// LiDARBEVCrossAttention_81071802679548
// MI455X (gfx1250) — matrix operand order repaired and checked by reading, not compiled, not run on hardware
//
#include <hip/hip_runtime.h>
#include <hip/hip_bf16.h>

typedef __attribute__((ext_vector_type(16))) _Float16 v16h;
typedef __attribute__((ext_vector_type(8)))  _Float16 v8h;
typedef __attribute__((ext_vector_type(8)))  float    v8f;

#define B_    2
#define NQ    1220
#define DQ    512
#define DKV   256
#define H0    512
#define W0    512
#define HP    128
#define WP    128
#define HWP   16384
#define KTOK  512
#define NH    8
#define HD    64
#define BN    (B_*NQ) // 2440

// ---------------------------------------------------------------------------
// K1: 4x4 average pool  pts[B,256,512,512] -> bev[B*256, 128*128]
// ---------------------------------------------------------------------------
__global__ void avgpool_kernel(const float* __restrict__ pts, float* __restrict__ bev) {
    int i = blockIdx.x * blockDim.x + threadIdx.x;
    if (i >= B_ * 256 * HWP) return;
    int x  = i & (WP - 1);
    int y  = (i >> 7) & (HP - 1);
    int bc = i >> 14;
    const float* p = pts + ((size_t)bc * H0 + (size_t)y * 4) * W0 + (size_t)x * 4;
    float s = 0.f;
#pragma unroll
    for (int dy = 0; dy < 4; ++dy) {
        const float* pr = p + (size_t)dy * W0;
        s += pr[0] + pr[1] + pr[2] + pr[3];
    }
    bev[i] = s * (1.f / 16.f);
}

// ---------------------------------------------------------------------------
// K2: score[b,p] = mean_c bev[b,c,p]^2
// ---------------------------------------------------------------------------
__global__ void score_kernel(const float* __restrict__ bev, float* __restrict__ score) {
    int i = blockIdx.x * blockDim.x + threadIdx.x;
    if (i >= B_ * HWP) return;
    int b = i >> 14, p = i & (HWP - 1);
    const float* base = bev + (size_t)b * 256 * HWP + p;
    float s = 0.f;
    for (int c = 0; c < 256; ++c) {
        float v = base[(size_t)c * HWP];
        s += v * v;
    }
    score[i] = s * (1.f / 256.f);
}

// ---------------------------------------------------------------------------
// K3: exact top-512 (stable descending, jax tie semantics) by rank counting.
// ---------------------------------------------------------------------------
__global__ void topk_kernel(const float* __restrict__ score, int* __restrict__ idxout) {
    extern __shared__ float s[];
    int b = blockIdx.y;
    int tid = threadIdx.x;
    for (int i = tid; i < HWP; i += 1024) s[i] = score[b * HWP + i];
    __syncthreads();
    int i = blockIdx.x * 1024 + tid;
    float vi = s[i];
    int rank = 0;
    for (int j = 0; j < HWP; ++j) {
        float vj = s[j];
        rank += (int)((vj > vi) || (vj == vi && j < i));
    }
    if (rank < KTOK) idxout[b * KTOK + rank] = i;
}

// ---------------------------------------------------------------------------
// K4: gather token, add bev_pos_mlp, layernorm(tn) -> kvln f16 [B*512, 256]
// ---------------------------------------------------------------------------
__global__ void token_kernel(const float* __restrict__ bev, const int* __restrict__ idx,
                             const float* __restrict__ w1, const float* __restrict__ b1,
                             const float* __restrict__ w2, const float* __restrict__ b2,
                             const float* __restrict__ tnw, const float* __restrict__ tnb,
                             _Float16* __restrict__ kvln) {
    __shared__ float h[256], r1[256], r2[256];
    int b = blockIdx.y, kt = blockIdx.x, tid = threadIdx.x;
    int p = idx[b * KTOK + kt];
    float px = ((float)(p & (WP - 1)) + 0.5f) * (1.f / (float)WP);
    float py = ((float)(p >> 7) + 0.5f) * (1.f / (float)HP);
    float tv = bev[((size_t)(b * 256 + tid)) * HWP + p];
    h[tid] = fmaxf(w1[tid * 2] * px + w1[tid * 2 + 1] * py + b1[tid], 0.f);
    __syncthreads();
    float val = tv + b2[tid];
    const float* w2r = w2 + (size_t)tid * 256;
    for (int j = 0; j < 256; ++j) val += h[j] * w2r[j];
    r1[tid] = val; r2[tid] = val * val;
    __syncthreads();
    for (int off = 128; off > 0; off >>= 1) {
        if (tid < off) { r1[tid] += r1[tid + off]; r2[tid] += r2[tid + off]; }
        __syncthreads();
    }
    float mean = r1[0] * (1.f / 256.f);
    float var  = r2[0] * (1.f / 256.f) - mean * mean;
    float rs   = rsqrtf(var + 1e-5f);
    kvln[((size_t)(b * KTOK + kt)) * 256 + tid] =
        (_Float16)((val - mean) * rs * tnw[tid] + tnb[tid]);
}

// ---------------------------------------------------------------------------
// K5: anchor positional relu features Hap f16 [BN,512]
// ---------------------------------------------------------------------------
__global__ void apfeat_kernel(const float* __restrict__ anchor, const float* __restrict__ w1,
                              const float* __restrict__ b1, _Float16* __restrict__ Hap) {
    int i = blockIdx.x * blockDim.x + threadIdx.x;
    if (i >= BN * DQ) return;
    int row = i >> 9, j = i & 511;
    float ax = anchor[(size_t)row * 11 + 0];
    float ay = anchor[(size_t)row * 11 + 1];
    float x = fminf(fmaxf((ax + 51.2f) * (1.f / 102.4f), 0.f), 1.f);
    float y = fminf(fmaxf((ay + 51.2f) * (1.f / 102.4f), 0.f), 1.f);
    Hap[i] = (_Float16)fmaxf(w1[j * 2] * x + w1[j * 2 + 1] * y + b1[j], 0.f);
}

// ---------------------------------------------------------------------------
// packw: W[N,K] f32 -> f16 fragment order (lane's 16 B values contiguous)
//   Wf[(((nt*ksteps)+ks)*32 + lane)*16 + e] = W[(nt*16+lane%16)*K + ks*32+16*(e/8)+8*(lane/16)+e%8]
// ---------------------------------------------------------------------------
__global__ void packw_kernel(const float* __restrict__ W, _Float16* __restrict__ Wf,
                             int N, int K) {
    int o = blockIdx.x * blockDim.x + threadIdx.x;
    if (o >= N * K) return;
    int e = o & 15;
    int lane = (o >> 4) & 31;
    int r = o >> 9;
    int ksteps = K >> 5;
    int ks = r % ksteps;
    int nt = r / ksteps;
    Wf[o] = (_Float16)W[(size_t)(nt * 16 + (lane & 15)) * K + (ks << 5) + ((e & 8) << 1) + ((lane >> 4) << 3) + (e & 7)];
}

// packk: K-proj [B*KTOK, 512] f32 -> per-(b,h) B-frag f16 for QK^T
//   gemm dims: n = token (32 tiles), k = d (2 ksteps of 32)
__global__ void packk_kernel(const float* __restrict__ kbuf, _Float16* __restrict__ Kf) {
    int o = blockIdx.x * blockDim.x + threadIdx.x;      // B*NH*KTOK*HD
    if (o >= B_ * NH * KTOK * HD) return;
    int oo = o & (KTOK * HD - 1);
    int bh = o >> 15;
    int b = bh >> 3, h = bh & 7;
    int e = oo & 15, lane = (oo >> 4) & 31, r = oo >> 9;
    int ks = r & 1, nt = r >> 1;
    int token = nt * 16 + (lane & 15), d = ks * 32 + ((e & 8) << 1) + ((lane >> 4) << 3) + (e & 7);
    Kf[o] = (_Float16)kbuf[((size_t)(b * KTOK + token)) * DQ + h * HD + d];
}

// packv: V-proj [B*KTOK, 512] f32 -> per-(b,h) B-frag f16 for PV
//   gemm dims: n = d (4 tiles), k = token (16 ksteps of 32)
__global__ void packv_kernel(const float* __restrict__ vbuf, _Float16* __restrict__ Vf) {
    int o = blockIdx.x * blockDim.x + threadIdx.x;
    if (o >= B_ * NH * KTOK * HD) return;
    int oo = o & (KTOK * HD - 1);
    int bh = o >> 15;
    int b = bh >> 3, h = bh & 7;
    int e = oo & 15, lane = (oo >> 4) & 31, r = oo >> 9;
    int ks = r & 15, nt = r >> 4;
    int token = ks * 32 + ((e & 8) << 1) + ((lane >> 4) << 3) + (e & 7), d = nt * 16 + (lane & 15);
    Vf[o] = (_Float16)vbuf[((size_t)(b * KTOK + token)) * DQ + h * HD + d];
}

// ---------------------------------------------------------------------------
// K6: WMMA GEMM  C[M,N] = epi(A[M,K] @ W^T + bias (+resid))
// ---------------------------------------------------------------------------
enum { EPI_BIAS = 0, EPI_RELU = 1, EPI_SIG = 2, EPI_RESID = 3 };

__device__ __forceinline__ v16h cat8(v8h lo, v8h hi) {
    return __builtin_shufflevector(lo, hi, 0, 1, 2, 3, 4, 5, 6, 7,
                                   8, 9, 10, 11, 12, 13, 14, 15);
}

template <int EPI, bool WF, bool WH>
__global__ void gemm_wmma_kernel(const _Float16* __restrict__ A,
                                 const _Float16* __restrict__ Wf,
                                 const float* __restrict__ bias,
                                 const float* __restrict__ resid,
                                 float* __restrict__ C, _Float16* __restrict__ Ch,
                                 int M, int N, int K) {
    int lane = threadIdx.x;             // block == 1 wave of 32
    int m0   = blockIdx.x * 32;
    int nt0  = blockIdx.y * 4;
    int half = lane >> 4;
    int lm   = lane & 15;
    int mA0 = m0 + lm;      if (mA0 >= M) mA0 = M - 1;
    int mA1 = m0 + 16 + lm; if (mA1 >= M) mA1 = M - 1;
    const _Float16* Ar0 = A + (size_t)mA0 * K;
    const _Float16* Ar1 = A + (size_t)mA1 * K;
    int ksteps = K >> 5;
    const _Float16* Wl = Wf + (size_t)lane * 16;

    v8f acc[2][4] = {};
    for (int ks = 0; ks < ksteps; ++ks) {
        int k0 = ks << 5;
        v16h a0 = cat8(*(const v8h*)(Ar0 + k0 + 8 * half),
                       *(const v8h*)(Ar0 + k0 + 16 + 8 * half));
        v16h a1 = cat8(*(const v8h*)(Ar1 + k0 + 8 * half),
                       *(const v8h*)(Ar1 + k0 + 16 + 8 * half));
#pragma unroll
        for (int t = 0; t < 4; ++t) {
            const _Float16* bp = Wl + ((size_t)((nt0 + t) * ksteps + ks) << 9);
            v16h bf = cat8(*(const v8h*)(bp), *(const v8h*)(bp + 8));
            acc[0][t] = __builtin_amdgcn_wmma_f32_16x16x32_f16(
                false, a0, false, bf, (short)0, acc[0][t], false, false);
            acc[1][t] = __builtin_amdgcn_wmma_f32_16x16x32_f16(
                false, a1, false, bf, (short)0, acc[1][t], false, false);
        }
    }
#pragma unroll
    for (int i = 0; i < 2; ++i) {
#pragma unroll
        for (int t = 0; t < 4; ++t) {
            int n = (nt0 + t) * 16 + lm;
            float bn = bias[n];
#pragma unroll
            for (int r = 0; r < 8; ++r) {
                int mm = m0 + 16 * i + r + 8 * half;
                if (mm < M) {
                    float val = acc[i][t][r] + bn;
                    if (EPI == EPI_RELU)       val = fmaxf(val, 0.f);
                    else if (EPI == EPI_SIG)   val = 1.f / (1.f + __expf(-val));
                    else if (EPI == EPI_RESID) val += resid[(size_t)mm * N + n];
                    if (WF) C[(size_t)mm * N + n]  = val;
                    if (WH) Ch[(size_t)mm * N + n] = (_Float16)val;
                }
            }
        }
    }
}

// ---------------------------------------------------------------------------
// K7: layernorm over 512, f16 output
// ---------------------------------------------------------------------------
__global__ void ln512h_kernel(const float* __restrict__ x, const float* __restrict__ w,
                              const float* __restrict__ g, _Float16* __restrict__ y) {
    __shared__ float r1[256], r2[256];
    int row = blockIdx.x, tid = threadIdx.x;
    const float* xr = x + (size_t)row * DQ;
    float a = xr[tid], c = xr[tid + 256];
    r1[tid] = a + c; r2[tid] = a * a + c * c;
    __syncthreads();
    for (int off = 128; off > 0; off >>= 1) {
        if (tid < off) { r1[tid] += r1[tid + off]; r2[tid] += r2[tid + off]; }
        __syncthreads();
    }
    float mean = r1[0] * (1.f / 512.f);
    float var  = r2[0] * (1.f / 512.f) - mean * mean;
    float rs   = rsqrtf(var + 1e-5f);
    size_t o = (size_t)row * DQ;
    y[o + tid]       = (_Float16)((a - mean) * rs * w[tid] + g[tid]);
    y[o + tid + 256] = (_Float16)((c - mean) * rs * w[tid + 256] + g[tid + 256]);
}

// ---------------------------------------------------------------------------
// K8: WMMA flash-style attention. One wave per 16-query tile per (b,h).
// Phase A: QK^T via WMMA -> 16x512 logits in LDS.
// Phase B: in-wave softmax (lane pair per row, shfl_xor across halves).
// Phase C: PV via WMMA with A fragments streamed from LDS (ds_load_b128).
// ---------------------------------------------------------------------------
__global__ void attn_wmma_kernel(const _Float16* __restrict__ qh,
                                 const _Float16* __restrict__ Kf,
                                 const _Float16* __restrict__ Vf,
                                 _Float16* __restrict__ ctx) {
    __shared__ float    att32[16][512];
    __shared__ _Float16 att16[16][512];
    int lane = threadIdx.x;
    int m0 = blockIdx.x * 16;
    int h = blockIdx.y, b = blockIdx.z;
    int lm = lane & 15, half = lane >> 4;

    int mq = m0 + lm; if (mq >= NQ) mq = NQ - 1;
    const _Float16* Qr = qh + ((size_t)(b * NQ + mq)) * DQ + h * HD;
    v16h aq0 = cat8(*(const v8h*)(Qr + 8 * half),      *(const v8h*)(Qr + 16 + 8 * half));
    v16h aq1 = cat8(*(const v8h*)(Qr + 32 + 8 * half), *(const v8h*)(Qr + 48 + 8 * half));

    const _Float16* Kb = Kf + ((size_t)(b * NH + h)) * (KTOK * HD) + lane * 16;
    for (int nt = 0; nt < 32; ++nt) {
        const _Float16* bp0 = Kb + ((size_t)(nt * 2) << 9);
        v16h bf0 = cat8(*(const v8h*)bp0, *(const v8h*)(bp0 + 8));
        v16h bf1 = cat8(*(const v8h*)(bp0 + 512), *(const v8h*)(bp0 + 520));
        v8f acc = {};
        acc = __builtin_amdgcn_wmma_f32_16x16x32_f16(false, aq0, false, bf0,
                                                     (short)0, acc, false, false);
        acc = __builtin_amdgcn_wmma_f32_16x16x32_f16(false, aq1, false, bf1,
                                                     (short)0, acc, false, false);
#pragma unroll
        for (int r = 0; r < 8; ++r)
            att32[r + 8 * half][nt * 16 + lm] = acc[r] * 0.125f;
    }
    __syncthreads();
    // softmax: lane (lm,half) owns row lm, columns [half*256, half*256+256)
    {
        const float* ar = &att32[lm][half * 256];
        float mx = -3.4e38f;
        for (int j = 0; j < 256; ++j) mx = fmaxf(mx, ar[j]);
        mx = fmaxf(mx, __shfl_xor(mx, 16, 32));
        float sum = 0.f;
        for (int j = 0; j < 256; ++j) sum += __expf(ar[j] - mx);
        sum += __shfl_xor(sum, 16, 32);
        float inv = 1.f / sum;
        _Float16* o16 = &att16[lm][half * 256];
        for (int j = 0; j < 256; ++j) o16[j] = (_Float16)(__expf(ar[j] - mx) * inv);
    }
    __syncthreads();
    // PV
    const _Float16* Vb = Vf + ((size_t)(b * NH + h)) * (KTOK * HD) + lane * 16;
    v8f acc[4] = {};
    for (int ks = 0; ks < 16; ++ks) {
        v16h af = cat8(*(const v8h*)(&att16[lm][ks * 32 + 8 * half]),
                       *(const v8h*)(&att16[lm][ks * 32 + 16 + 8 * half]));
#pragma unroll
        for (int t = 0; t < 4; ++t) {
            const _Float16* bp = Vb + ((size_t)(t * 16 + ks) << 9);
            v16h bf = cat8(*(const v8h*)bp, *(const v8h*)(bp + 8));
            acc[t] = __builtin_amdgcn_wmma_f32_16x16x32_f16(
                false, af, false, bf, (short)0, acc[t], false, false);
        }
    }
#pragma unroll
    for (int t = 0; t < 4; ++t) {
        int d = t * 16 + lm;
#pragma unroll
        for (int r = 0; r < 8; ++r) {
            int mm = m0 + r + 8 * half;
            if (mm < NQ)
                ctx[((size_t)(b * NQ + mm)) * DQ + h * HD + d] = (_Float16)acc[t][r];
        }
    }
}

// ---------------------------------------------------------------------------
// K9: out = instance + gate * LN(preln; on_w,on_b)
// ---------------------------------------------------------------------------
__global__ void final_kernel(const float* __restrict__ preln, const float* __restrict__ onw,
                             const float* __restrict__ onb, const float* __restrict__ gate,
                             const float* __restrict__ inst, float* __restrict__ out) {
    __shared__ float r1[256], r2[256];
    int row = blockIdx.x, tid = threadIdx.x;
    const float* xr = preln + (size_t)row * DQ;
    float a = xr[tid], c = xr[tid + 256];
    r1[tid] = a + c; r2[tid] = a * a + c * c;
    __syncthreads();
    for (int off = 128; off > 0; off >>= 1) {
        if (tid < off) { r1[tid] += r1[tid + off]; r2[tid] += r2[tid + off]; }
        __syncthreads();
    }
    float mean = r1[0] * (1.f / 512.f);
    float var  = r2[0] * (1.f / 512.f) - mean * mean;
    float rs   = rsqrtf(var + 1e-5f);
    size_t o = (size_t)row * DQ;
    float l0 = (a - mean) * rs * onw[tid] + onb[tid];
    float l1 = (c - mean) * rs * onw[tid + 256] + onb[tid + 256];
    out[o + tid]       = inst[o + tid]       + gate[o + tid]       * l0;
    out[o + tid + 256] = inst[o + tid + 256] + gate[o + tid + 256] * l1;
}

// ---------------------------------------------------------------------------
extern "C" void kernel_launch(void* const* d_in, const int* in_sizes, int n_in,
                              void* d_out, int out_size, void* d_ws, size_t ws_size,
                              hipStream_t stream) {
    const float* inst   = (const float*)d_in[0];
    const float* pts    = (const float*)d_in[1];
    const float* anchor = (const float*)d_in[2];
    const float* qw  = (const float*)d_in[3];  const float* qb  = (const float*)d_in[4];
    const float* kw  = (const float*)d_in[5];  const float* kb  = (const float*)d_in[6];
    const float* vw  = (const float*)d_in[7];  const float* vb  = (const float*)d_in[8];
    const float* aow = (const float*)d_in[9];  const float* aob = (const float*)d_in[10];
    const float* qnw = (const float*)d_in[11]; const float* qnb = (const float*)d_in[12];
    const float* tnw = (const float*)d_in[13]; const float* tnb = (const float*)d_in[14];
    const float* onw = (const float*)d_in[15]; const float* onb = (const float*)d_in[16];
    const float* opw = (const float*)d_in[17]; const float* opb = (const float*)d_in[18];
    const float* bpw1 = (const float*)d_in[19]; const float* bpb1 = (const float*)d_in[20];
    const float* bpw2 = (const float*)d_in[21]; const float* bpb2 = (const float*)d_in[22];
    const float* apw1 = (const float*)d_in[23]; const float* apb1 = (const float*)d_in[24];
    const float* apw2 = (const float*)d_in[25]; const float* apb2 = (const float*)d_in[26];
    const float* gw1  = (const float*)d_in[27]; const float* gb1  = (const float*)d_in[28];
    const float* gw2  = (const float*)d_in[29]; const float* gb2  = (const float*)d_in[30];
    float* out = (float*)d_out;
    float* f = (float*)d_ws;

    // ---- workspace layout (units of float) ----
    // Region A: bev [8,388,608] recycled after token_kernel.
    float*     bev   = f + 0;
    float*     qi    = f + 0;                            // [BN,512] f32
    float*     gateb = f + 1249280;                      // [BN,512] f32
    float*     preln = f + 2498560;                      // [BN,512] f32
    _Float16*  qih   = (_Float16*)(f + 3747840);         // [BN,512] f16
    _Float16*  g1h   = (_Float16*)(f + 4372480);         // [BN,512] f16
    _Float16*  qlnh  = (_Float16*)(f + 4997120);         // [BN,512] f16
    _Float16*  ctxh  = (_Float16*)(f + 5621760);         // [BN,512] f16
    _Float16*  ctx2h = (_Float16*)(f + 6246400);         // [BN,512] f16 (ends 6,871,040)
    // Region B
    float*     score = f + 8388608;                      // [B,16384]
    int*       idxb  = (int*)(f + 8421376);              // [B,512]
    _Float16*  kvlnh = (_Float16*)(f + 8422400);         // [B*512,256] f16
    _Float16*  Haph  = (_Float16*)(f + 8553472);         // [BN,512] f16
    _Float16*  qh    = (_Float16*)(f + 9178112);         // [BN,512] f16
    float*     kbuf  = f + 9802752;                      // [B*512,512] f32
    float*     vbuf  = f + 10327040;                     // [B*512,512] f32
    _Float16*  Kf    = (_Float16*)(f + 10851328);        // [B*NH,512,64] frag f16
    _Float16*  Vf    = (_Float16*)(f + 11113472);        // [B*NH,512,64] frag f16
    _Float16*  qwf   = (_Float16*)(f + 11375616);        // frag-packed weights
    _Float16*  kwf   = (_Float16*)(f + 11506688);
    _Float16*  vwf   = (_Float16*)(f + 11572224);
    _Float16*  aowf  = (_Float16*)(f + 11637760);
    _Float16*  opwf  = (_Float16*)(f + 11768832);
    _Float16*  apw2f = (_Float16*)(f + 11899904);
    _Float16*  gw1f  = (_Float16*)(f + 12030976);
    _Float16*  gw2f  = (_Float16*)(f + 12162048);        // ends 12,293,120 floats

    // ---- weight fragment packing ----
    int nk512 = DQ * DQ, nk256 = DQ * DKV;
    packw_kernel<<<(nk512 + 255) / 256, 256, 0, stream>>>(qw,   qwf,   DQ, DQ);
    packw_kernel<<<(nk256 + 255) / 256, 256, 0, stream>>>(kw,   kwf,   DQ, DKV);
    packw_kernel<<<(nk256 + 255) / 256, 256, 0, stream>>>(vw,   vwf,   DQ, DKV);
    packw_kernel<<<(nk512 + 255) / 256, 256, 0, stream>>>(aow,  aowf,  DQ, DQ);
    packw_kernel<<<(nk512 + 255) / 256, 256, 0, stream>>>(opw,  opwf,  DQ, DQ);
    packw_kernel<<<(nk512 + 255) / 256, 256, 0, stream>>>(apw2, apw2f, DQ, DQ);
    packw_kernel<<<(nk512 + 255) / 256, 256, 0, stream>>>(gw1,  gw1f,  DQ, DQ);
    packw_kernel<<<(nk512 + 255) / 256, 256, 0, stream>>>(gw2,  gw2f,  DQ, DQ);

    // ---- front end ----
    avgpool_kernel<<<(B_ * 256 * HWP + 255) / 256, 256, 0, stream>>>(pts, bev);
    score_kernel<<<(B_ * HWP + 255) / 256, 256, 0, stream>>>(bev, score);
    topk_kernel<<<dim3(16, B_), 1024, HWP * sizeof(float), stream>>>(score, idxb);
    token_kernel<<<dim3(KTOK, B_), 256, 0, stream>>>(bev, idxb, bpw1, bpb1, bpw2, bpb2,
                                                     tnw, tnb, kvlnh);
    apfeat_kernel<<<(BN * DQ + 255) / 256, 256, 0, stream>>>(anchor, apw1, apb1, Haph);

    dim3 gBig((BN + 31) / 32, DQ / 64);
    dim3 gKV((B_ * KTOK + 31) / 32, DQ / 64);
    // query_input = instance + Hap @ apw2^T + apb2
    gemm_wmma_kernel<EPI_RESID, true, true><<<gBig, 32, 0, stream>>>(
        Haph, apw2f, apb2, inst, qi, qih, BN, DQ, DQ);
    // g1 = relu(qi @ gw1^T + gb1)
    gemm_wmma_kernel<EPI_RELU, false, true><<<gBig, 32, 0, stream>>>(
        qih, gw1f, gb1, nullptr, nullptr, g1h, BN, DQ, DQ);
    // gate = sigmoid(g1 @ gw2^T + gb2)
    gemm_wmma_kernel<EPI_SIG, true, false><<<gBig, 32, 0, stream>>>(
        g1h, gw2f, gb2, nullptr, gateb, nullptr, BN, DQ, DQ);
    // qln = LN(qi; qn)
    ln512h_kernel<<<BN, 256, 0, stream>>>(qi, qnw, qnb, qlnh);
    // q projection -> f16 row-major only; k/v -> f32 for frag packing
    gemm_wmma_kernel<EPI_BIAS, false, true><<<gBig, 32, 0, stream>>>(
        qlnh, qwf, qb, nullptr, nullptr, qh, BN, DQ, DQ);
    gemm_wmma_kernel<EPI_BIAS, true, false><<<gKV, 32, 0, stream>>>(
        kvlnh, kwf, kb, nullptr, kbuf, nullptr, B_ * KTOK, DQ, DKV);
    gemm_wmma_kernel<EPI_BIAS, true, false><<<gKV, 32, 0, stream>>>(
        kvlnh, vwf, vb, nullptr, vbuf, nullptr, B_ * KTOK, DQ, DKV);
    // pack K/V into per-(b,h) WMMA B fragments
    int nkv = B_ * NH * KTOK * HD;
    packk_kernel<<<(nkv + 255) / 256, 256, 0, stream>>>(kbuf, Kf);
    packv_kernel<<<(nkv + 255) / 256, 256, 0, stream>>>(vbuf, Vf);
    // WMMA attention -> f16 ctx
    attn_wmma_kernel<<<dim3((NQ + 15) / 16, NH, B_), 32, 0, stream>>>(qh, Kf, Vf, ctxh);
    // MHA out_proj, then module out_proj
    gemm_wmma_kernel<EPI_BIAS, false, true><<<gBig, 32, 0, stream>>>(
        ctxh, aowf, aob, nullptr, nullptr, ctx2h, BN, DQ, DQ);
    gemm_wmma_kernel<EPI_BIAS, true, false><<<gBig, 32, 0, stream>>>(
        ctx2h, opwf, opb, nullptr, preln, nullptr, BN, DQ, DQ);
    // out = instance + gate * LN(preln; on)
    final_kernel<<<BN, 256, 0, stream>>>(preln, onw, onb, gateb, inst, out);

    (void)in_sizes; (void)n_in; (void)out_size; (void)ws_size;
}
